// HeterogeneousMoE_52338471469233
// MI455X (gfx1250) — matrix operand order repaired and checked by reading, not compiled, not run on hardware
//
#include <hip/hip_runtime.h>

// ---- problem constants (match reference) ----
#define T_LEN 256
#define NTOK  4096          // T*B
#define DDIM  512
#define NTASK 4

// ---- CDNA5 async-to-LDS path (guarded; falls back to global reads) ----
#ifndef __has_builtin
#define __has_builtin(x) 0
#endif
#if __has_builtin(__builtin_amdgcn_global_load_async_to_lds_b32) && \
    __has_builtin(__builtin_amdgcn_s_wait_asynccnt)
#define HAVE_ASYNC_LDS 1
#else
#define HAVE_ASYNC_LDS 0
#endif
#define AS1 __attribute__((address_space(1)))
#define AS3 __attribute__((address_space(3)))

typedef __attribute__((ext_vector_type(16))) __bf16        v16bf;
typedef __attribute__((ext_vector_type(8)))  float         v8f;
typedef __attribute__((ext_vector_type(8)))  unsigned int  v8u;

// round-to-nearest-even f32 -> bf16 bits
__device__ inline unsigned short f2bf(float f) {
  union { float f; unsigned int u; } c; c.f = f;
  unsigned int u = c.u;
  u += 0x7fffu + ((u >> 16) & 1u);
  return (unsigned short)(u >> 16);
}

// ---------------- fragment loaders (wave32 WMMA layouts, ISA 7.12.2) -------
__device__ inline v16bf load_fragA(const unsigned short* __restrict__ base,
                                   int row, int ld, int kbase, int lane) {
  int hi = (lane >> 4) & 1;
  const unsigned short* p = base + (size_t)row * ld + kbase + hi * 8;
  v8u u;
#pragma unroll
  for (int v = 0; v < 8; ++v) {
    int k = (v < 4) ? (2 * v) : (16 + 2 * (v - 4));
    u[v] = *(const unsigned int*)(p + k);
  }
  return __builtin_bit_cast(v16bf, u);
}

__device__ inline v16bf load_fragB(const unsigned short* __restrict__ base,
                                   int row, int ld, int kbase, int lane) {
  int hi = (lane >> 4) & 1;
  const unsigned short* p = base + (size_t)row * ld + kbase + hi * 8;
  v8u u;
#pragma unroll
  for (int v = 0; v < 8; ++v) u[v] = *(const unsigned int*)(p + ((v < 4) ? (2 * v) : (16 + 2 * (v - 4))));
  return __builtin_bit_cast(v16bf, u);
}

#define WMMA_BF16(A_, B_, C_) \
  __builtin_amdgcn_wmma_f32_16x16x32_bf16(false, (A_), false, (B_), (short)0, (C_), false, false)

// ---------------- generic bf16 WMMA GEMM:  C[m,n] = sum_k A[m,k]*W[n,k] ----
// One wave computes a 32x32 tile (2x2 WMMA) with a software-pipelined K loop:
// next chunk's 4 fragments are in flight while 4 WMMAs consume the current one.
// mode 0: C = A@W^T + bias
// mode 1: out[m,n] += gates[m,expert] * (A@W^T + bias)[m,n]   (Ncols==DDIM)
__global__ void __launch_bounds__(256)
wmma_gemm(const unsigned short* __restrict__ A,   // M x K  bf16
          const unsigned short* __restrict__ W,   // Ncols x K bf16
          const float* __restrict__ bias,         // Ncols
          float* __restrict__ C,
          float* __restrict__ outAcc,
          const float* __restrict__ gates,        // NTOK x 4
          int M, int Ncols, int K, int mode, int expert)
{
  int lane  = threadIdx.x & 31;
  int wave  = blockIdx.x * (blockDim.x >> 5) + (threadIdx.x >> 5);
  int tilesN = Ncols >> 5;                         // 32-wide tiles
  int tilesM = M >> 5;
  if (wave >= tilesM * tilesN) return;             // whole-wave exit: EXEC all-ones
  int tm = wave / tilesN, tn = wave % tilesN;
  int arow0 = tm * 32 + (lane & 15), arow1 = arow0 + 16;
  int brow0 = tn * 32 + (lane & 15), brow1 = brow0 + 16;
  const unsigned short* pa0 = A + (size_t)arow0 * K;

  v16bf a0 = load_fragA(A, arow0, K, 0, lane);
  v16bf a1 = load_fragA(A, arow1, K, 0, lane);
  v16bf b0 = load_fragB(W, brow0, K, 0, lane);
  v16bf b1 = load_fragB(W, brow1, K, 0, lane);
  v8f acc00 = {}, acc01 = {}, acc10 = {}, acc11 = {};

  for (int k0 = 32; k0 < K; k0 += 32) {
    __builtin_prefetch(pa0 + k0 + 32, 0, 1);       // global_prefetch_b8
    v16bf na0 = load_fragA(A, arow0, K, k0, lane);
    v16bf na1 = load_fragA(A, arow1, K, k0, lane);
    v16bf nb0 = load_fragB(W, brow0, K, k0, lane);
    v16bf nb1 = load_fragB(W, brow1, K, k0, lane);
    acc00 = WMMA_BF16(a0, b0, acc00);
    acc01 = WMMA_BF16(a0, b1, acc01);
    acc10 = WMMA_BF16(a1, b0, acc10);
    acc11 = WMMA_BF16(a1, b1, acc11);
    a0 = na0; a1 = na1; b0 = nb0; b1 = nb1;
  }
  acc00 = WMMA_BF16(a0, b0, acc00);
  acc01 = WMMA_BF16(a0, b1, acc01);
  acc10 = WMMA_BF16(a1, b0, acc10);
  acc11 = WMMA_BF16(a1, b1, acc11);

  int hi = (lane >> 4) & 1;
  int c0 = tn * 32 + (lane & 15), c1 = c0 + 16;
  float bc0 = bias ? bias[c0] : 0.0f;
  float bc1 = bias ? bias[c1] : 0.0f;
#pragma unroll
  for (int r = 0; r < 8; ++r) {
    int r0 = tm * 32 + r + 8 * hi, r1 = r0 + 16;
    float v00 = acc00[r] + bc0, v01 = acc01[r] + bc1;
    float v10 = acc10[r] + bc0, v11 = acc11[r] + bc1;
    if (mode == 0) {
      C[(size_t)r0 * Ncols + c0] = v00;  C[(size_t)r0 * Ncols + c1] = v01;
      C[(size_t)r1 * Ncols + c0] = v10;  C[(size_t)r1 * Ncols + c1] = v11;
    } else {
      float g0 = gates[r0 * 4 + expert], g1 = gates[r1 * 4 + expert];
      outAcc[(size_t)r0 * DDIM + c0] += g0 * v00;
      outAcc[(size_t)r0 * DDIM + c1] += g0 * v01;
      outAcc[(size_t)r1 * DDIM + c0] += g1 * v10;
      outAcc[(size_t)r1 * DDIM + c1] += g1 * v11;
    }
  }
}

// ---------------- f32 -> bf16 conversion --------------------------------
__global__ void cvt_bf16(const float* __restrict__ in, unsigned short* __restrict__ out, int n) {
  int i = blockIdx.x * blockDim.x + threadIdx.x;
  if (i < n) out[i] = f2bf(in[i]);
}

// ---------------- router recurrence (persistent, 16 waves, WMMA per step) --
__global__ void __launch_bounds__(512)
router_recur(const float* __restrict__ gi,           // NTOK x 768 (b_ih folded in)
             const unsigned short* __restrict__ whh, // 768 x 256 bf16 row-major
             const float* __restrict__ bhh,          // 768
             float* __restrict__ hs)                 // NTOK x 256
{
  extern __shared__ float smem[];
  float* h  = smem;                 // 16*256
  float* gh = smem + 16 * 256;      // 16*768
  int tid = threadIdx.x;
  int lane = tid & 31, wave = tid >> 5;

  for (int i = tid; i < 16 * 256; i += 512) h[i] = 0.0f;
  __syncthreads();

  for (int t = 0; t < T_LEN; ++t) {
    // gh = h @ whh^T : M=16,K=256,N=768 -> 48 tiles over 16 waves
#pragma unroll
    for (int s = 0; s < 3; ++s) {
      int tn = wave * 3 + s;
      int brow = tn * 16 + (lane & 15);
      v8f acc = {};
      for (int k0 = 0; k0 < 256; k0 += 32) {
        int m = lane & 15, hA = (lane >> 4) & 1;
        v8u u;
#pragma unroll
        for (int v = 0; v < 8; ++v) {
          int kl = ((v < 4) ? (2 * v) : (16 + 2 * (v - 4))) + hA * 8;
          float f0 = h[m * 256 + k0 + kl];
          float f1 = h[m * 256 + k0 + kl + 1];
          u[v] = (unsigned int)f2bf(f0) | ((unsigned int)f2bf(f1) << 16);
        }
        v16bf a = __builtin_bit_cast(v16bf, u);
        v16bf b = load_fragB(whh, brow, 256, k0, lane);
        acc = WMMA_BF16(a, b, acc);
      }
      int hi = (lane >> 4) & 1, col = tn * 16 + (lane & 15);
#pragma unroll
      for (int r = 0; r < 8; ++r) gh[(r + 8 * hi) * 768 + col] = acc[r];
    }
    __syncthreads();

    for (int i = tid; i < 16 * 256; i += 512) {
      int b = i >> 8, k = i & 255;
      const float* git = gi + (size_t)(t * 16 + b) * 768;
      float ghr = gh[b * 768 + k]       + bhh[k];
      float ghz = gh[b * 768 + 256 + k] + bhh[256 + k];
      float ghn = gh[b * 768 + 512 + k] + bhh[512 + k];
      float r = 1.0f / (1.0f + __expf(-(git[k] + ghr)));
      float z = 1.0f / (1.0f + __expf(-(git[256 + k] + ghz)));
      float nn = tanhf(git[512 + k] + r * ghn);
      float hn = (1.0f - z) * nn + z * h[i];
      h[i] = hn;
      hs[(size_t)(t * 16 + b) * 256 + k] = hn;
    }
    __syncthreads();
  }
}

// ---------------- router head: relu(hs)@routw^T, softmax, task atomics ----
__global__ void router_head(const float* __restrict__ hs,
                            const float* __restrict__ rw,   // 4 x 256
                            const float* __restrict__ rb,   // 4
                            const int*   __restrict__ task_ids,
                            float* __restrict__ raw,        // NTOK x 4 (in d_out)
                            float* __restrict__ taskAcc)    // sums[4], cnts[4]
{
  int gw   = (blockIdx.x * blockDim.x + threadIdx.x) >> 5;  // one wave per token
  int lane = threadIdx.x & 31;
  if (gw >= NTOK) return;
  const float* hrow = hs + (size_t)gw * 256;
  float p[4] = {0.f, 0.f, 0.f, 0.f};
  for (int j = lane; j < 256; j += 32) {
    float hv = fmaxf(hrow[j], 0.0f);
#pragma unroll
    for (int e = 0; e < 4; ++e) p[e] += hv * rw[e * 256 + j];
  }
#pragma unroll
  for (int off = 16; off > 0; off >>= 1)
#pragma unroll
    for (int e = 0; e < 4; ++e) p[e] += __shfl_down(p[e], off, 32);
  if (lane == 0) {
    float mx = -1e30f;
#pragma unroll
    for (int e = 0; e < 4; ++e) { p[e] += rb[e]; mx = fmaxf(mx, p[e]); }
    float s = 0.f;
#pragma unroll
    for (int e = 0; e < 4; ++e) { p[e] = __expf(p[e] - mx); s += p[e]; }
    float inv = 1.0f / s, eul = 0.f;
    const float cost[4] = {0.f, 128.f, 256.f, 512.f};
#pragma unroll
    for (int e = 0; e < 4; ++e) { p[e] *= inv; raw[gw * 4 + e] = p[e]; eul += p[e] * cost[e]; }
    int tsk = task_ids[gw] & 3;
    atomicAdd(&taskAcc[tsk], eul);
    atomicAdd(&taskAcc[4 + tsk], 1.0f);
  }
}

__global__ void zero_f32(float* p, int n) {
  int i = blockIdx.x * blockDim.x + threadIdx.x;
  if (i < n) p[i] = 0.0f;
}

__global__ void task_losses(const float* __restrict__ acc, float* __restrict__ out) {
  int e = threadIdx.x;
  if (e < NTASK) { float c = acc[4 + e]; out[e] = (c > 0.f) ? acc[e] / c : 0.f; }
}

// expert 0 is identity: out = gates[:,0] * x
__global__ void init_out(const float* __restrict__ x, const float* __restrict__ raw,
                         float* __restrict__ out) {
  int i = blockIdx.x * blockDim.x + threadIdx.x;
  if (i < NTOK * DDIM) out[i] = raw[(i >> 9) * 4] * x[i];
}

// ---------------- expert pre-transpose:  whhT[k][j] = whh[j][k] -----------
__global__ void transpose_f32(const float* __restrict__ in, float* __restrict__ out,
                              int rows, int cols) {
  int i = blockIdx.x * blockDim.x + threadIdx.x;
  if (i < rows * cols) { int r = i / cols, c = i % cols; out[(size_t)c * rows + r] = in[i]; }
}

// ---------------- expert recurrence: unbatched GRU over NTOK serial steps --
// h in LDS; per step gh = b_hh + h @ whhT (coalesced, L2-resident weights).
// CDNA5 path: gi row for step n+1 is async-copied global->LDS (ASYNCcnt)
// while the GEMV for step n runs; double-buffered, published via barrier.
__global__ void __launch_bounds__(1024)
expert_recur(const float* __restrict__ gi,    // NTOK x 3H (b_ih folded in)
             const float* __restrict__ whhT,  // H x 3H
             const float* __restrict__ bhh,   // 3H
             float* __restrict__ hout,        // NTOK x H
             int H)
{
  __shared__ float h[512];
  __shared__ float gh[1536];
#if HAVE_ASYNC_LDS
  __shared__ float gibuf[2][1536];
#endif
  int tid = threadIdx.x;
  int H3 = 3 * H;
  for (int i = tid; i < H; i += blockDim.x) h[i] = 0.0f;
#if HAVE_ASYNC_LDS
  for (int j = tid; j < H3; j += blockDim.x)
    __builtin_amdgcn_global_load_async_to_lds_b32(
        (AS1 int*)(void*)(gi + j), (AS3 int*)&gibuf[0][j], 0, 0);
  __builtin_amdgcn_s_wait_asynccnt(0);
#endif
  __syncthreads();

  for (int n = 0; n < NTOK; ++n) {
#if HAVE_ASYNC_LDS
    if (n + 1 < NTOK)
      for (int j = tid; j < H3; j += blockDim.x)
        __builtin_amdgcn_global_load_async_to_lds_b32(
            (AS1 int*)(void*)(gi + (size_t)(n + 1) * H3 + j),
            (AS3 int*)&gibuf[(n + 1) & 1][j], 0, 0);
    const float* gin = gibuf[n & 1];
#else
    const float* gin = gi + (size_t)n * H3;
#endif
    for (int j = tid; j < H3; j += blockDim.x) {
      float acc = bhh[j];
      const float* wp = whhT + j;
      for (int k = 0; k < H; k += 4) {
        acc += h[k]     * wp[(size_t)(k)     * H3];
        acc += h[k + 1] * wp[(size_t)(k + 1) * H3];
        acc += h[k + 2] * wp[(size_t)(k + 2) * H3];
        acc += h[k + 3] * wp[(size_t)(k + 3) * H3];
      }
      gh[j] = acc;
    }
    __syncthreads();
    for (int k = tid; k < H; k += blockDim.x) {
      float r  = 1.0f / (1.0f + __expf(-(gin[k] + gh[k])));
      float z  = 1.0f / (1.0f + __expf(-(gin[H + k] + gh[H + k])));
      float nn = tanhf(gin[2 * H + k] + r * gh[2 * H + k]);
      float hn = (1.0f - z) * nn + z * h[k];
      h[k] = hn;
      hout[(size_t)n * H + k] = hn;
    }
#if HAVE_ASYNC_LDS
    __builtin_amdgcn_s_wait_asynccnt(0);
#endif
    __syncthreads();
  }
}

// ---------------- BN stats (train-mode, biased var) -> scale/shift --------
__global__ void bn_stats(const float* __restrict__ hbuf, const float* __restrict__ g,
                         const float* __restrict__ b, float* __restrict__ scale,
                         float* __restrict__ shift, int H)
{
  int j = blockIdx.x, tid = threadIdx.x;
  __shared__ float s1[256], s2[256];
  float a1 = 0.f, a2 = 0.f;
  for (int n = tid; n < NTOK; n += 256) {
    float v = hbuf[(size_t)n * H + j];
    a1 += v; a2 += v * v;
  }
  s1[tid] = a1; s2[tid] = a2; __syncthreads();
  for (int s = 128; s > 0; s >>= 1) {
    if (tid < s) { s1[tid] += s1[tid + s]; s2[tid] += s2[tid + s]; }
    __syncthreads();
  }
  if (tid == 0) {
    float mu   = s1[0] / (float)NTOK;
    float var  = s2[0] / (float)NTOK - mu * mu;
    float rstd = rsqrtf(var + 1e-5f);
    float sc   = g[j] * rstd;
    scale[j] = sc; shift[j] = b[j] - mu * sc;
  }
}

// fused BN-affine + ReLU + bf16 pack for the WMMA output projection
__global__ void bn_relu_bf16(const float* __restrict__ hbuf, const float* __restrict__ scale,
                             const float* __restrict__ shift, unsigned short* __restrict__ outbf,
                             int H)
{
  int i = blockIdx.x * blockDim.x + threadIdx.x;
  if (i < NTOK * H) {
    int j = i & (H - 1);
    outbf[i] = f2bf(fmaxf(hbuf[i] * scale[j] + shift[j], 0.0f));
  }
}

// =========================================================================
extern "C" void kernel_launch(void* const* d_in, const int* in_sizes, int n_in,
                              void* d_out, int out_size, void* d_ws, size_t ws_size,
                              hipStream_t stream)
{
  (void)in_sizes; (void)n_in; (void)out_size; (void)ws_size;

  const float* x     = (const float*)d_in[0];
  const int*   tids  = (const int*)d_in[1];
  const float* rwih  = (const float*)d_in[2];
  const float* rwhh  = (const float*)d_in[3];
  const float* rbih  = (const float*)d_in[4];
  const float* rbhh  = (const float*)d_in[5];
  const float* routw = (const float*)d_in[6];
  const float* routb = (const float*)d_in[7];

  const float *e_wih[3], *e_whh[3], *e_bih[3], *e_bhh[3], *e_bng[3], *e_bnb[3], *e_ow[3], *e_ob[3];
  for (int i = 0; i < 3; ++i) {
    int b = 8 + i * 8;
    e_wih[i] = (const float*)d_in[b + 0];
    e_whh[i] = (const float*)d_in[b + 1];
    e_bih[i] = (const float*)d_in[b + 2];
    e_bhh[i] = (const float*)d_in[b + 3];
    e_bng[i] = (const float*)d_in[b + 4];
    e_bnb[i] = (const float*)d_in[b + 5];
    e_ow[i]  = (const float*)d_in[b + 6];
    e_ob[i]  = (const float*)d_in[b + 7];
  }
  const int EH[3] = {128, 256, 512};

  // ---- workspace bump allocator (~67 MB total) ----
  char* ws = (char*)d_ws;
  size_t off = 0;
  auto alloc = [&](size_t bytes) -> void* {
    void* p = ws + off;
    off = (off + bytes + 255) & ~(size_t)255;
    return p;
  };
  unsigned short* x_bf    = (unsigned short*)alloc((size_t)NTOK * DDIM * 2);
  unsigned short* rwih_bf = (unsigned short*)alloc((size_t)768 * 512 * 2);
  unsigned short* rwhh_bf = (unsigned short*)alloc((size_t)768 * 256 * 2);
  unsigned short* ewih_bf = (unsigned short*)alloc((size_t)2688 * 512 * 2); // 384+768+1536 rows
  unsigned short* eow_bf  = (unsigned short*)alloc((size_t)512 * 896 * 2);  // 128+256+512 cols
  float* gi_r   = (float*)alloc((size_t)NTOK * 768 * 4);
  float* hsbuf  = (float*)alloc((size_t)NTOK * 256 * 4);
  float* gi_e   = (float*)alloc((size_t)NTOK * 1536 * 4);   // reused per expert
  float* hbuf   = (float*)alloc((size_t)NTOK * 512 * 4);    // reused
  unsigned short* hnbf = (unsigned short*)alloc((size_t)NTOK * 512 * 2); // reused
  float* whhT   = (float*)alloc((size_t)512 * 1536 * 4);    // reused
  float* bnsc   = (float*)alloc(512 * 4);
  float* bnsh   = (float*)alloc(512 * 4);
  float* tacc   = (float*)alloc(8 * 4);

  float* out  = (float*)d_out;
  float* rawp = out + (size_t)NTOK * DDIM;    // raw router output (== gates)
  float* tlp  = rawp + (size_t)NTOK * 4;      // task losses (4)

  auto cvt = [&](const float* in, unsigned short* o, int n) {
    cvt_bf16<<<(n + 255) / 256, 256, 0, stream>>>(in, o, n);
  };
  auto gemm = [&](const unsigned short* A, const unsigned short* W, const float* bias,
                  float* C, float* outAcc, const float* gates,
                  int M, int Nc, int K, int mode, int expert) {
    int tiles  = (M >> 5) * (Nc >> 5);          // 32x32 tile per wave
    int blocks = (tiles + 7) / 8;               // 8 waves / 256 threads per block
    wmma_gemm<<<blocks, 256, 0, stream>>>(A, W, bias, C, outAcc, gates,
                                          M, Nc, K, mode, expert);
  };

  // ---- bf16 staging of activations + all GEMM weights ----
  cvt(x, x_bf, NTOK * DDIM);
  cvt(rwih, rwih_bf, 768 * 512);
  cvt(rwhh, rwhh_bf, 768 * 256);
  const int ewoff[3] = {0, 384 * 512, (384 + 768) * 512};
  const int eooff[3] = {0, 512 * 128, 512 * (128 + 256)};
  for (int i = 0; i < 3; ++i) {
    cvt(e_wih[i], ewih_bf + ewoff[i], 3 * EH[i] * 512);
    cvt(e_ow[i],  eow_bf  + eooff[i], 512 * EH[i]);
  }

  // ---- router: gi GEMM -> persistent WMMA recurrence -> head/softmax ----
  gemm(x_bf, rwih_bf, rbih, gi_r, nullptr, nullptr, NTOK, 768, 512, 0, 0);
  router_recur<<<1, 512, (16 * 256 + 16 * 768) * sizeof(float), stream>>>(
      gi_r, rwhh_bf, rbhh, hsbuf);
  zero_f32<<<1, 32, 0, stream>>>(tacc, 8);
  router_head<<<(NTOK * 32) / 256, 256, 0, stream>>>(hsbuf, routw, routb, tids, rawp, tacc);
  task_losses<<<1, NTASK, 0, stream>>>(tacc, tlp);

  // ---- expert 0 (identity) initializes out ----
  init_out<<<(NTOK * DDIM + 255) / 256, 256, 0, stream>>>(x, rawp, out);

  // ---- experts 1..3 ----
  for (int i = 0; i < 3; ++i) {
    int H = EH[i], H3 = 3 * H;
    gemm(x_bf, ewih_bf + ewoff[i], e_bih[i], gi_e, nullptr, nullptr, NTOK, H3, 512, 0, 0);
    transpose_f32<<<(H3 * H + 255) / 256, 256, 0, stream>>>(e_whh[i], whhT, H3, H);
    expert_recur<<<1, 1024, 0, stream>>>(gi_e, whhT, e_bhh[i], hbuf, H);
    bn_stats<<<H, 256, 0, stream>>>(hbuf, e_bng[i], e_bnb[i], bnsc, bnsh, H);
    bn_relu_bf16<<<(NTOK * H + 255) / 256, 256, 0, stream>>>(hbuf, bnsc, bnsh, hnbf, H);
    // fused: out += gate_i * (relu(bn(h)) @ ow^T + ob)
    gemm(hnbf, eow_bf + eooff[i], e_ob[i], nullptr, out, rawp, NTOK, DDIM, H, 1, i + 1);
  }
}
